// WeightedMultiHeadAttention_36361193128578
// MI455X (gfx1250) — compile-verified, matrix operand order checked by reading, not run on hardware
//
#include <hip/hip_runtime.h>
#include <hip/hip_bf16.h>
#include <stdint.h>

// ---------------------------------------------------------------------------
// Weighted multi-head attention for MI455X (gfx1250), wave32 + WMMA bf16
// with async global->LDS tile DMA (ASYNCcnt) and double-buffered GEMM tiles.
//  Pipeline:
//   1) cvt_f32_bf16 x3        : qb/kb/vb   = bf16(q/k/v)
//   2) transpose_cvt x4       : W*t[n][k]  = bf16(W*[k][n])
//   3) gemm_async<bf16> x3    : Qp/Kp/Vp   = qb@Wq+b (bf16)
//   4) attn_kernel            : Ctx[b,s,h*64+d] (bf16)
//   5) gemm_async<float>      : out        = Ctx@Wo+bo (f32)
// ---------------------------------------------------------------------------

typedef __bf16 bf16;
typedef __attribute__((ext_vector_type(16))) __bf16 v16bf;
typedef __attribute__((ext_vector_type(4)))  __bf16 v4bf;
typedef __attribute__((ext_vector_type(8)))  float  v8f;
typedef __attribute__((ext_vector_type(4)))  int    v4i;

union FragBF { v16bf v; v4i q[2]; };

// Load one wave32 16-bit WMMA fragment from an LDS tile.
// Layout (ISA 7.12.2): lane holds 8 contiguous elements at kb and 8 at kb+16.
static __device__ __forceinline__ v16bf ld_frag(const bf16* p) {
  FragBF f;
  f.q[0] = *(const v4i*)(p);
  f.q[1] = *(const v4i*)(p + 16);
  return f.v;
}

static __device__ __forceinline__ v8f wmma_bf16(v16bf a, v16bf b, v8f c) {
  return __builtin_amdgcn_wmma_f32_16x16x32_bf16(
      false, a, false, b, (short)0, c, false, false);
}

// Async global->LDS 16B copy (GLOBAL_LOAD_ASYNC_TO_LDS_B128, ASYNCcnt).
// lds_off = byte offset into wave's LDS (low 32 bits of generic shared addr).
static __device__ __forceinline__ void async_ld_b128(uint32_t lds_off,
                                                     const bf16* g) {
  asm volatile("global_load_async_to_lds_b128 %0, %1, off"
               :: "v"(lds_off), "v"(g)
               : "memory");
}

static __device__ __forceinline__ uint32_t lds_addr(const void* p) {
  return (uint32_t)(uintptr_t)p;  // LDS aperture: addr[31:0] = LDS offset
}

// ---------------------------------------------------------------------------
// Elementwise f32 -> bf16 (vector of 4 per thread)
// ---------------------------------------------------------------------------
__global__ void __launch_bounds__(256)
cvt_f32_bf16(const float* __restrict__ in, bf16* __restrict__ out, int n) {
  int i = (blockIdx.x * 256 + threadIdx.x) * 4;
  if (i >= n) return;
  const float4 f = *(const float4*)(in + i);
  v4bf t;
  t[0] = (bf16)f.x; t[1] = (bf16)f.y; t[2] = (bf16)f.z; t[3] = (bf16)f.w;
  *(v4bf*)(out + i) = t;
}

// ---------------------------------------------------------------------------
// Wt[n][k] = bf16(W[k][n]); 32x32 LDS tile transpose, block 256 = 32x8
// ---------------------------------------------------------------------------
__global__ void __launch_bounds__(256)
transpose_cvt(const float* __restrict__ W, bf16* __restrict__ Wt,
              int K, int N) {
  __shared__ float t[32][33];
  const int k0 = blockIdx.y * 32, n0 = blockIdx.x * 32;
  const int tx = threadIdx.x & 31, ty = threadIdx.x >> 5;  // 32 x 8
#pragma unroll
  for (int j = 0; j < 4; ++j)
    t[ty + 8 * j][tx] = W[(size_t)(k0 + ty + 8 * j) * N + n0 + tx];
  __syncthreads();
#pragma unroll
  for (int j = 0; j < 4; ++j)
    Wt[(size_t)(n0 + ty + 8 * j) * K + k0 + tx] = (bf16)t[tx][ty + 8 * j];
}

// ---------------------------------------------------------------------------
// GEMM: C[M,N] = A[M,K](bf16) @ Bt[N,K](bf16, pre-transposed) + bias[N]
//   block 256 (8 waves), tile 128x128, K-step 32, wave grid 4x2 (32x64 each).
//   Double-buffered LDS tiles filled by async global->LDS B128 DMA:
//   4 async instructions per wave per stage -> s_wait_asynccnt 4 overlaps
//   stage i+1 DMA with stage i WMMAs.
// ---------------------------------------------------------------------------
template <typename TO>
__global__ void __launch_bounds__(256)
gemm_async(const bf16* __restrict__ A, const bf16* __restrict__ Bt,
           const float* __restrict__ bias, TO* __restrict__ C,
           int M, int K, int N) {
  constexpr int BM = 128, BN = 128, BK = 32, LDP = 40;
  __shared__ bf16 As[2][BM * LDP];
  __shared__ bf16 Bs[2][BN * LDP];

  const int tid  = threadIdx.x;
  const int m0   = blockIdx.y * BM;
  const int n0   = blockIdx.x * BN;
  const int w    = tid >> 5;
  const int lane = tid & 31;
  const int lo   = lane & 15;
  const int hi   = lane >> 4;
  const int wr   = w >> 1;  // 0..3 : 32-row band
  const int wc   = w & 1;   // 0..1 : 64-col band

  const uint32_t asb[2] = {lds_addr(&As[0][0]), lds_addr(&As[1][0])};
  const uint32_t bsb[2] = {lds_addr(&Bs[0][0]), lds_addr(&Bs[1][0])};

  // Issue one stage: 512 B128 chunks per tile, 2 per thread per tile.
  auto issue = [&](int k0, int buf) {
#pragma unroll
    for (int j = 0; j < 2; ++j) {
      const int i = tid + j * 256;       // 0..511
      const int r = i >> 2;              // row 0..127
      const int c = i & 3;               // 8-elem chunk 0..3
      const uint32_t loff = (uint32_t)(r * LDP + c * 8) * 2;
      async_ld_b128(asb[buf] + loff, A  + (size_t)(m0 + r) * K + k0 + c * 8);
      async_ld_b128(bsb[buf] + loff, Bt + (size_t)(n0 + r) * K + k0 + c * 8);
    }
  };

  v8f acc[2][4] = {};
  const int nstages = K / BK;

  issue(0, 0);
  for (int s = 0; s < nstages; ++s) {
    const int cur = s & 1;
    if (s + 1 < nstages) {
      issue((s + 1) * BK, cur ^ 1);
      asm volatile("s_wait_asynccnt 0x4" ::: "memory");  // stage s landed
    } else {
      asm volatile("s_wait_asynccnt 0x0" ::: "memory");
    }
    __syncthreads();  // all waves' DMA for stage s visible

    FragBF aF[2];
#pragma unroll
    for (int ms = 0; ms < 2; ++ms)
      aF[ms].v = ld_frag(&As[cur][(wr * 32 + ms * 16 + lo) * LDP + hi * 8]);
#pragma unroll
    for (int ns = 0; ns < 4; ++ns) {
      v16bf bF = ld_frag(&Bs[cur][(wc * 64 + ns * 16 + lo) * LDP + hi * 8]);
#pragma unroll
      for (int ms = 0; ms < 2; ++ms)
        acc[ms][ns] = wmma_bf16(aF[ms].v, bF, acc[ms][ns]);
    }
    __syncthreads();  // buffer cur may be overwritten by stage s+2
  }

  // epilogue: +bias, store. C layout: row = v + 8*hi, col = lo
#pragma unroll
  for (int ms = 0; ms < 2; ++ms) {
#pragma unroll
    for (int ns = 0; ns < 4; ++ns) {
      const int col = n0 + wc * 64 + ns * 16 + lo;
      const float bv = bias[col];
#pragma unroll
      for (int v = 0; v < 8; ++v) {
        const int row = m0 + wr * 32 + ms * 16 + v + 8 * hi;
        C[(size_t)row * N + col] = (TO)(acc[ms][ns][v] + bv);
      }
    }
  }
}

// ---------------------------------------------------------------------------
// Attention: one block per (b, h, 64-row q tile); 4 waves, 16 q-rows each.
//   S = Q K^T / 8 -> p = exp(S) * w[k]; rowsum via shfl_xor; ctx += P V;
//   final ctx / (rowsum + 1e-12). Q/K tiles staged by async global->LDS DMA.
// ---------------------------------------------------------------------------
__global__ void __launch_bounds__(128)
attn_kernel(const bf16* __restrict__ Qp, const bf16* __restrict__ Kp,
            const bf16* __restrict__ Vp, const float* __restrict__ wvec,
            bf16* __restrict__ Ctx) {
  constexpr int S = 2048, D = 1024, Dh = 64, QT = 64, KT = 64, LDP = 72;
  __shared__ bf16 Qs[QT * LDP];      // Qs[q][d]
  __shared__ bf16 Ks[KT * LDP];      // Ks[k][d]  (acts as B-transposed)
  __shared__ bf16 Vt[Dh * LDP];      // Vt[d][k]  (V transposed)
  __shared__ bf16 Ps[4 * 16 * LDP];  // per-wave P block [16 q][64 k]

  const int tid   = threadIdx.x;
  const int b     = blockIdx.y >> 4;  // / H
  const int h     = blockIdx.y & 15;  // % H
  const int qBase = blockIdx.x * QT;
  const int w     = tid >> 5;
  const int lane  = tid & 31;
  const int lo    = lane & 15;
  const int hi    = lane >> 4;

  const uint32_t qsb = lds_addr(&Qs[0]);
  const uint32_t ksb = lds_addr(&Ks[0]);

  // stage Q tile once (async DMA): 512 B128 chunks, 4 per thread
#pragma unroll
  for (int j = 0; j < 4; ++j) {
    const int i = tid + j * 128;
    const int r = i >> 3, c = i & 7;
    async_ld_b128(qsb + (uint32_t)(r * LDP + c * 8) * 2,
                  Qp + ((size_t)b * S + qBase + r) * D + h * Dh + c * 8);
  }

  v8f   ctx[4]    = {};
  float rowsum[8] = {0, 0, 0, 0, 0, 0, 0, 0};

  for (int kt = 0; kt < S; kt += KT) {
    __syncthreads();  // previous iteration done reading Ks/Vt
    // K tile: async DMA
#pragma unroll
    for (int j = 0; j < 4; ++j) {
      const int i = tid + j * 128;
      const int r = i >> 3, c = i & 7;
      async_ld_b128(ksb + (uint32_t)(r * LDP + c * 8) * 2,
                    Kp + ((size_t)b * S + kt + r) * D + h * Dh + c * 8);
    }
    // V tile transposed (synchronous scatter)
    for (int i = tid; i < (KT * Dh) / 4; i += 128) {
      const int r = i >> 4, c = (i & 15) << 2;
      const bf16* src = Vp + ((size_t)b * S + kt + r) * D + h * Dh + c;
#pragma unroll
      for (int j = 0; j < 4; ++j) Vt[(c + j) * LDP + r] = src[j];
    }
    asm volatile("s_wait_asynccnt 0x0" ::: "memory");
    __syncthreads();

    // scores: 16 q-rows (this wave) x 64 k-cols, contracted over Dh=64
    v8f sacc[4] = {};
#pragma unroll
    for (int ks = 0; ks < 2; ++ks) {
      v16bf aq = ld_frag(&Qs[(w * 16 + lo) * LDP + ks * 32 + hi * 8]);
#pragma unroll
      for (int n = 0; n < 4; ++n) {
        v16bf bk = ld_frag(&Ks[(n * 16 + lo) * LDP + ks * 32 + hi * 8]);
        sacc[n] = wmma_bf16(aq, bk, sacc[n]);
      }
    }

    // p = exp(s / sqrt(64)) * w[k]; accumulate row sums; spill P to LDS
    float part[8] = {0, 0, 0, 0, 0, 0, 0, 0};
#pragma unroll
    for (int n = 0; n < 4; ++n) {
      const float wv = wvec[(size_t)b * S + kt + n * 16 + lo];
#pragma unroll
      for (int v = 0; v < 8; ++v) {
        const float p = __expf(sacc[n][v] * 0.125f) * wv;
        part[v] += p;
        Ps[(w * 16 + v + 8 * hi) * LDP + n * 16 + lo] = (bf16)p;
      }
    }
#pragma unroll
    for (int m = 1; m < 16; m <<= 1)
#pragma unroll
      for (int v = 0; v < 8; ++v) part[v] += __shfl_xor(part[v], m, 32);
#pragma unroll
    for (int v = 0; v < 8; ++v) rowsum[v] += part[v];

    __syncthreads();  // make P visible (conservative; P is wave-private)

    // ctx += P @ V
#pragma unroll
    for (int ks = 0; ks < 2; ++ks) {
      v16bf ap = ld_frag(&Ps[(w * 16 + lo) * LDP + ks * 32 + hi * 8]);
#pragma unroll
      for (int n = 0; n < 4; ++n) {
        v16bf bv = ld_frag(&Vt[(n * 16 + lo) * LDP + ks * 32 + hi * 8]);
        ctx[n] = wmma_bf16(ap, bv, ctx[n]);
      }
    }
  }

  // normalize and store: Ctx[b, qBase+r, h*64+d]
#pragma unroll
  for (int n = 0; n < 4; ++n) {
    const int d = n * 16 + lo;
#pragma unroll
    for (int v = 0; v < 8; ++v) {
      const int r = w * 16 + v + 8 * hi;
      const float val = ctx[n][v] / (rowsum[v] + 1e-12f);
      Ctx[((size_t)b * S + qBase + r) * D + h * Dh + d] = (bf16)val;
    }
  }
}

// ---------------------------------------------------------------------------
extern "C" void kernel_launch(void* const* d_in, const int* in_sizes, int n_in,
                              void* d_out, int out_size, void* d_ws,
                              size_t ws_size, hipStream_t stream) {
  constexpr int B = 4, S = 2048, D = 1024, H = 16;
  const int M = B * S, K = D, N = D;
  const size_t MD = (size_t)M * D;   // 8M elems
  const size_t KN = (size_t)K * N;   // 1M elems

  const float* q  = (const float*)d_in[0];
  const float* k  = (const float*)d_in[1];
  const float* v  = (const float*)d_in[2];
  const float* wv = (const float*)d_in[3];
  const float* Wq = (const float*)d_in[4];
  const float* bq = (const float*)d_in[5];
  const float* Wk = (const float*)d_in[6];
  const float* bk = (const float*)d_in[7];
  const float* Wv = (const float*)d_in[8];
  const float* bv = (const float*)d_in[9];
  const float* Wo = (const float*)d_in[10];
  const float* bo = (const float*)d_in[11];

  // workspace layout (bf16): 7 x 16MB activations + 4 x 2MB weights = 120 MB
  bf16* qb  = (bf16*)d_ws;
  bf16* kb  = qb + MD;
  bf16* vb  = kb + MD;
  bf16* Qp  = vb + MD;
  bf16* Kp  = Qp + MD;
  bf16* Vp  = Kp + MD;
  bf16* Ctx = Vp + MD;
  bf16* Wqt = Ctx + MD;
  bf16* Wkt = Wqt + KN;
  bf16* Wvt = Wkt + KN;
  bf16* Wot = Wvt + KN;

  // 1) activations -> bf16
  const int nEl = (int)MD;
  cvt_f32_bf16<<<nEl / 1024, 256, 0, stream>>>(q, qb, nEl);
  cvt_f32_bf16<<<nEl / 1024, 256, 0, stream>>>(k, kb, nEl);
  cvt_f32_bf16<<<nEl / 1024, 256, 0, stream>>>(v, vb, nEl);

  // 2) weights -> bf16 transposed [N][K]
  dim3 tgrid(N / 32, K / 32);
  transpose_cvt<<<tgrid, 256, 0, stream>>>(Wq, Wqt, K, N);
  transpose_cvt<<<tgrid, 256, 0, stream>>>(Wk, Wkt, K, N);
  transpose_cvt<<<tgrid, 256, 0, stream>>>(Wv, Wvt, K, N);
  transpose_cvt<<<tgrid, 256, 0, stream>>>(Wo, Wot, K, N);

  // 3) projections
  dim3 ggrid(N / 128, M / 128);  // 8 x 64
  gemm_async<bf16><<<ggrid, 256, 0, stream>>>(qb, Wqt, bq, Qp, M, K, N);
  gemm_async<bf16><<<ggrid, 256, 0, stream>>>(kb, Wkt, bk, Kp, M, K, N);
  gemm_async<bf16><<<ggrid, 256, 0, stream>>>(vb, Wvt, bv, Vp, M, K, N);

  // 4) attention
  attn_kernel<<<dim3(S / 64, B * H), 128, 0, stream>>>(Qp, Kp, Vp, wv, Ctx);

  // 5) output projection (f32 out)
  gemm_async<float><<<ggrid, 256, 0, stream>>>(Ctx, Wot, bo,
                                               (float*)d_out, M, K, N);
}
